// BiLSTMLM_63848983822342
// MI455X (gfx1250) — compile-verified, matrix operand order checked by reading, not run on hardware
//
#include <hip/hip_runtime.h>
#include <hip/hip_bf16.h>

typedef __attribute__((ext_vector_type(16))) _Float16 v16h;
typedef __attribute__((ext_vector_type(8)))  float    v8f;

// Load one 32-element f16 row in the WMMA 16-bit A/B per-lane packing:
// lanes 0-15 (s=0): K = {0..7, 16..23}; lanes 16-31 (s=1): K = {8..15, 24..31}.
__device__ inline v16h load_tile_row(const _Float16* __restrict__ row, int s) {
    union { v16h h; uint4 q[2]; } u;
    u.q[0] = *(const uint4*)(row + 8 * s);
    u.q[1] = *(const uint4*)(row + 16 + 8 * s);
    return u.h;
}

// ---------------------------------------------------------------------------
// Kernel 1: W_out f32 -> f16 (row-major [V,32]); L2-resident thereafter.
// ---------------------------------------------------------------------------
__global__ void cvt_f32_to_f16_kernel(const float* __restrict__ x,
                                      _Float16* __restrict__ y, int n) {
    int i = blockIdx.x * blockDim.x + threadIdx.x;
    if (i < n) y[i] = (_Float16)x[i];
}

// ---------------------------------------------------------------------------
// Kernel 2: bidirectional LSTM scan (exact f32). blockIdx.x = direction.
// Writes A[S*16][32] f16: A[w][0:16] = hLR_used[w], A[w][16:32] = hRL_used[w].
// Pre-update h written at index w is correct for both directions.
// ---------------------------------------------------------------------------
__global__ __launch_bounds__(256) void bilstm_kernel(
    const int*   __restrict__ tok,   const float* __restrict__ emb,
    const float* __restrict__ W_ih,  const float* __restrict__ W_hh,
    const float* __restrict__ b_ih,  const float* __restrict__ b_hh,
    const float* __restrict__ h0,    const float* __restrict__ c0,
    _Float16*    __restrict__ A,     int S) {
    __shared__ float sWih[64][32];
    __shared__ float sWhh[64][16];
    __shared__ float sb[64];
    __shared__ float sx[16][32];
    __shared__ float sh[16][16];

    const int tid = threadIdx.x;
    const int b = tid >> 4, j = tid & 15;
    for (int i = tid; i < 64 * 32; i += 256) sWih[i >> 5][i & 31] = W_ih[i];
    for (int i = tid; i < 64 * 16; i += 256) sWhh[i >> 4][i & 15] = W_hh[i];
    if (tid < 64) sb[tid] = b_ih[tid] + b_hh[tid];

    float h = h0[j], c = c0[j];
    sh[b][j] = h;
    const int dir = blockIdx.x;
    __syncthreads();

    for (int t = 0; t < S; ++t) {
        const int w = dir ? (S - 1 - t) : t;
        // pre-update hidden is exactly the context vector used at position w
        A[((size_t)w * 16 + b) * 32 + dir * 16 + j] = (_Float16)h;
        // stage x = embedding[token[w, bb]] for all 16 batches
        for (int p = tid; p < 16 * 32; p += 256) {
            int bb = p >> 5, e = p & 31;
            sx[bb][e] = emb[(size_t)tok[w * 16 + bb] * 32 + e];
        }
        __syncthreads();
        float gi = sb[j], gf = sb[16 + j], gg = sb[32 + j], go = sb[48 + j];
#pragma unroll
        for (int e = 0; e < 32; ++e) {
            float xv = sx[b][e];
            gi += sWih[j][e] * xv;      gf += sWih[16 + j][e] * xv;
            gg += sWih[32 + j][e] * xv; go += sWih[48 + j][e] * xv;
        }
#pragma unroll
        for (int k = 0; k < 16; ++k) {
            float hv = sh[b][k];
            gi += sWhh[j][k] * hv;      gf += sWhh[16 + j][k] * hv;
            gg += sWhh[32 + j][k] * hv; go += sWhh[48 + j][k] * hv;
        }
        float si = 1.0f / (1.0f + __expf(-gi));
        float sf = 1.0f / (1.0f + __expf(-gf));
        float so = 1.0f / (1.0f + __expf(-go));
        float cn = sf * c + si * tanhf(gg);
        float hn = so * tanhf(cn);
        __syncthreads();          // all reads of old sh/sx done
        h = hn; c = cn; sh[b][j] = hn;
    }
}

// ---------------------------------------------------------------------------
// Kernel 3: fused [16,32]x[32,V] WMMA GEMM + online log-softmax, recompute
// pass 2 (A tile stays in 8 VGPRs, W_out f16 re-read from L2).
// One block per 16-row tile; 8 waves split the V/16 column tiles.
// ---------------------------------------------------------------------------
__global__ __launch_bounds__(256) void gemm_logsoftmax_kernel(
    const _Float16* __restrict__ A, const _Float16* __restrict__ Wf,
    const float* __restrict__ b_out, float* __restrict__ out,
    int ntiles, int V) {
    __shared__ float redM[8][16];
    __shared__ float redS[8][16];
    __shared__ float rowc[16];

    const int tid  = threadIdx.x;
    const int wave = tid >> 5, lane = tid & 31;
    const int l16  = lane & 15, s = lane >> 4;
    const int rt   = blockIdx.x;

    const v16h a = load_tile_row(A + (size_t)(rt * 16 + l16) * 32, s);

    float m[8], sm[8];
#pragma unroll
    for (int r = 0; r < 8; ++r) { m[r] = -3.4e38f; sm[r] = 0.0f; }

    // ---- pass 1: per-lane online max / sum-exp over this wave's columns ----
    for (int nt = wave; nt < ntiles; nt += 8) {
        const int col = nt * 16 + l16;
        v16h bv = load_tile_row(Wf + (size_t)col * 32, s);
        v8f  cacc = {};
        cacc = __builtin_amdgcn_wmma_f32_16x16x32_f16(
            false, a, false, bv, (short)0, cacc, false, false);
        const float bias = b_out[col];
#pragma unroll
        for (int r = 0; r < 8; ++r) {
            float v  = cacc[r] + bias;
            float nm = fmaxf(m[r], v);
            sm[r] = sm[r] * __expf(m[r] - nm) + __expf(v - nm);
            m[r]  = nm;
        }
    }
    // merge (m, s) across the 16 lanes of each half-wave
#pragma unroll
    for (int r = 0; r < 8; ++r) {
#pragma unroll
        for (int msk = 1; msk < 16; msk <<= 1) {
            float om = __shfl_xor(m[r],  msk, 32);
            float os = __shfl_xor(sm[r], msk, 32);
            float nm = fmaxf(m[r], om);
            sm[r] = sm[r] * __expf(m[r] - nm) + os * __expf(om - nm);
            m[r]  = nm;
        }
    }
    if (l16 == 0) {
#pragma unroll
        for (int r = 0; r < 8; ++r) {
            redM[wave][8 * s + r] = m[r];
            redS[wave][8 * s + r] = sm[r];
        }
    }
    __syncthreads();
    if (tid < 16) {  // combine the 8 waves' partials per row
        float M = -3.4e38f;
        for (int w = 0; w < 8; ++w) M = fmaxf(M, redM[w][tid]);
        float S = 0.0f;
        for (int w = 0; w < 8; ++w) S += redS[w][tid] * __expf(redM[w][tid] - M);
        rowc[tid] = M + __logf(S);
    }
    __syncthreads();
    float rc[8];
#pragma unroll
    for (int r = 0; r < 8; ++r) rc[r] = rowc[8 * s + r];

    // ---- pass 2: recompute logits, write log-softmax straight to HBM ----
    for (int nt = wave; nt < ntiles; nt += 8) {
        const int col = nt * 16 + l16;
        v16h bv = load_tile_row(Wf + (size_t)col * 32, s);
        v8f  cacc = {};
        cacc = __builtin_amdgcn_wmma_f32_16x16x32_f16(
            false, a, false, bv, (short)0, cacc, false, false);
        const float bias = b_out[col];
#pragma unroll
        for (int r = 0; r < 8; ++r) {
            float v = cacc[r] + bias;
            out[(size_t)(rt * 16 + 8 * s + r) * V + col] = v - rc[r];
        }
    }
}

extern "C" void kernel_launch(void* const* d_in, const int* in_sizes, int n_in,
                              void* d_out, int out_size, void* d_ws, size_t ws_size,
                              hipStream_t stream) {
    const int*   tok   = (const int*)d_in[0];
    const float* emb   = (const float*)d_in[1];
    const float* W_ih  = (const float*)d_in[2];
    const float* W_hh  = (const float*)d_in[3];
    const float* b_ih  = (const float*)d_in[4];
    const float* b_hh  = (const float*)d_in[5];
    const float* W_out = (const float*)d_in[6];
    const float* b_out = (const float*)d_in[7];
    const float* h0    = (const float*)d_in[8];
    const float* c0    = (const float*)d_in[9];
    float* out = (float*)d_out;

    const int SB = in_sizes[0];   // S*B = 4096
    const int S  = SB / 16;       // 256
    const int V  = in_sizes[7];   // 32000

    _Float16* Af = (_Float16*)d_ws;                                  // [SB][32]
    _Float16* Wf = (_Float16*)((char*)d_ws + (size_t)SB * 32 * 2);   // [V][32]

    const int nW = V * 32;
    cvt_f32_to_f16_kernel<<<(nW + 255) / 256, 256, 0, stream>>>(W_out, Wf, nW);
    bilstm_kernel<<<2, 256, 0, stream>>>(tok, emb, W_ih, W_hh, b_ih, b_hh,
                                         h0, c0, Af, S);
    gemm_logsoftmax_kernel<<<SB / 16, 256, 0, stream>>>(Af, Wf, b_out, out,
                                                        V / 16, V);
}
